// Head_1580547975322
// MI455X (gfx1250) — matrix operand order repaired and checked by reading, not compiled, not run on hardware
//
#include <hip/hip_runtime.h>

typedef __attribute__((ext_vector_type(16))) __bf16 bf16x16;
typedef __attribute__((ext_vector_type(8)))  __bf16 bf16x8;
typedef __attribute__((ext_vector_type(8)))  float  f32x8;

#define B_ 32
#define T_ 1536
#define C_ 512
#define H_ 64

__device__ __forceinline__ f32x8 wmma_bf16(bf16x16 a, bf16x16 b, f32x8 c) {
  // 8 args: (neg_a, A, neg_b, B, c_mod, C, reuse_a, reuse_b)
  return __builtin_amdgcn_wmma_f32_16x16x32_bf16(false, a, false, b, (short)0, c,
                                                 false, false);
}

// A fragment (16x32 bf16): lane m=lane&15 holds row m.
//   a[0..7]  = row[16*half + i],  a[8..15] = row[16*half + 8 + (i-8)]
__device__ __forceinline__ bf16x16 a_frag(const __bf16* row, int half) {
  bf16x16 a;
  const __bf16* p0 = row + 16*half;
  const __bf16* p1 = row + 16*half + 8;
#pragma unroll
  for (int i = 0; i < 8; ++i) { a[i] = p0[i]; a[i+8] = p1[i]; }
  return a;
}

// B fragment (32x16): b[i] = B[K = i + 16*half][N = lane&15],
// with storage arranged so K is the contiguous axis -> 2 x b128 per lane.
__device__ __forceinline__ bf16x16 b_frag_contig(const __bf16* p) {
  bf16x16 b;
#pragma unroll
  for (int i = 0; i < 16; ++i) b[i] = p[i];
  return b;
}

// async global->LDS copy of 16 bytes per lane (gfx1250 ASYNCcnt path)
__device__ __forceinline__ void async_b128(uint32_t lds_off, const void* g) {
  asm volatile("global_load_async_to_lds_b128 %0, %1, off"
               :: "v"(lds_off), "v"((unsigned long long)(uintptr_t)g)
               : "memory");
}
__device__ __forceinline__ void wait_async0() {
  asm volatile("s_wait_asynccnt 0x0" ::: "memory");
}

// ---------------- kernel 0: W -> bf16, transposed to [H][C] ----------------
__global__ __launch_bounds__(256) void cvt_w_kernel(const float* __restrict__ wq,
                                                    const float* __restrict__ wk,
                                                    const float* __restrict__ wv,
                                                    __bf16* __restrict__ o) {
  int i = blockIdx.x * 256 + threadIdx.x;     // over C_*H_, c-major read (coalesced)
  if (i < C_*H_) {
    int c = i >> 6, h = i & 63;
    o[h*C_ + c]             = (__bf16)wq[i];
    o[C_*H_ + h*C_ + c]     = (__bf16)wk[i];
    o[2*C_*H_ + h*C_ + c]   = (__bf16)wv[i];
  }
}

// ---------------- kernel 1: QKV projection ----------------
// q,k: row-major [B*T][H].  v: transposed vT [B][H][T].
__global__ __launch_bounds__(128) void qkv_proj_kernel(
    const float* __restrict__ x,
    const __bf16* __restrict__ wqt, const __bf16* __restrict__ wkt,
    const __bf16* __restrict__ wvt,
    __bf16* __restrict__ q, __bf16* __restrict__ k, __bf16* __restrict__ vT)
{
  __shared__ __bf16 xs[16][C_ + 8];   // row stride 1040B = 65*16: aligned + bank-spread
  const int tid = threadIdx.x;
  const size_t row0 = (size_t)blockIdx.x * 16;

  // stage 16x512 f32 -> bf16 into LDS (float4 loads)
#pragma unroll 4
  for (int e = tid; e < 16 * (C_/4); e += 128) {
    int r = e >> 7;
    int c = (e & 127) << 2;
    const float4 f = *(const float4*)(x + (row0 + r)*C_ + c);
    xs[r][c+0] = (__bf16)f.x;
    xs[r][c+1] = (__bf16)f.y;
    xs[r][c+2] = (__bf16)f.z;
    xs[r][c+3] = (__bf16)f.w;
  }
  __syncthreads();

  const int wave = tid >> 5, lane = tid & 31;
  const int half = lane >> 4, n = lane & 15;
  const int hb = wave * 16;                 // this wave's 16-wide h tile
  const size_t wrow = (size_t)(hb + n) * C_;
  f32x8 cq = {}, ck = {}, cv = {};

  // one-stage software pipeline: fetch kk+32 fragments before current wmmas
  bf16x16 a_c  = a_frag(&xs[n][0], half);
  bf16x16 bq_c = b_frag_contig(wqt + wrow + 16*half);
  bf16x16 bk_c = b_frag_contig(wkt + wrow + 16*half);
  bf16x16 bv_c = b_frag_contig(wvt + wrow + 16*half);
#pragma unroll 4
  for (int kk = 0; kk < C_; kk += 32) {
    const int kn   = (kk + 32) & (C_ - 1);  // wraps on last iter (harmless reload)
    const int kofs = kn + 16*half;
    bf16x16 a_n  = a_frag(&xs[n][kn], half);
    bf16x16 bq_n = b_frag_contig(wqt + wrow + kofs);
    bf16x16 bk_n = b_frag_contig(wkt + wrow + kofs);
    bf16x16 bv_n = b_frag_contig(wvt + wrow + kofs);
    cq = wmma_bf16(a_c, bq_c, cq);
    ck = wmma_bf16(a_c, bk_c, ck);
    cv = wmma_bf16(a_c, bv_c, cv);
    a_c = a_n; bq_c = bq_n; bk_c = bk_n; bv_c = bv_n;
  }

  // q,k row-major stores (C/D layout: VGPR r -> row r+8*half)
#pragma unroll
  for (int r = 0; r < 8; ++r) {
    const size_t row = row0 + r + 8*half;
    q[row*H_ + hb + n] = (__bf16)cq[r];
    k[row*H_ + hb + n] = (__bf16)ck[r];
  }
  // v transposed store: lane holds 8 consecutive tokens for column h -> one b128
  {
    const int bidx = (int)(row0 / T_);      // 16-row tiles never straddle a batch
    const int t0   = (int)(row0 % T_);
    bf16x8 pv;
#pragma unroll
    for (int r = 0; r < 8; ++r) pv[r] = (__bf16)cv[r];
    *(bf16x8*)(vT + ((size_t)bidx*H_ + hb + n)*T_ + t0 + 8*half) = pv;
  }
}

// ---------------- kernel 2: causal flash attention ----------------
__global__ __launch_bounds__(128) void attn_kernel(
    const __bf16* __restrict__ q, const __bf16* __restrict__ k,
    const __bf16* __restrict__ vT, float* __restrict__ out)
{
  __shared__ __bf16 kt[2][32][H_ + 8];  // K tile, row-major [s][h], stride 144B
  __shared__ __bf16 vt[2][H_][32 + 8];  // V tile, transposed [h][s], stride 80B
  __shared__ __bf16 pt[4][16][40];      // per-wave P staging (C-frag -> A-frag)

  const int b  = blockIdx.y;
  const int qb = blockIdx.x;                         // 64 query rows per block
  const int tid = threadIdx.x, wave = tid >> 5, lane = tid & 31;
  const int half = lane >> 4, n = lane & 15;

  const size_t qrow0 = (size_t)b * T_ + qb*64 + wave*16;
  const bf16x16 aq0 = a_frag(q + (qrow0 + n)*H_ +  0, half);   // Q K-chunk 0..31
  const bf16x16 aq1 = a_frag(q + (qrow0 + n)*H_ + 32, half);   // Q K-chunk 32..63

  f32x8 o[4] = {};
  float mrow[8], lrow[8];
#pragma unroll
  for (int r = 0; r < 8; ++r) { mrow[r] = -1e30f; lrow[r] = 0.f; }

  const int jmax = qb*2 + 2;            // causal: 32-key blocks up to q-block end

  // async stage of key-block j into LDS buffer `buf` (4 x b128 per thread)
  auto stage = [&](int j, int buf) {
    {  // K tile: 32 rows x 64 h, row-major
      const int r = tid >> 2, cs = (tid & 3) * 16;
      const __bf16* g = k + ((size_t)b*T_ + j*32 + r)*H_ + cs;
      const uint32_t l = (uint32_t)(uintptr_t)&kt[buf][r][cs];
      async_b128(l,      g);
      async_b128(l + 16, g + 8);
    }
    {  // V tile: 64 h-rows x 32 s, from vT (already transposed in global)
      const int hr = tid >> 1, cs = (tid & 1) * 16;
      const __bf16* g = vT + ((size_t)b*H_ + hr)*T_ + j*32 + cs;
      const uint32_t l = (uint32_t)(uintptr_t)&vt[buf][hr][cs];
      async_b128(l,      g);
      async_b128(l + 16, g + 8);
    }
  };

  stage(0, 0);
  for (int j = 0; j < jmax; ++j) {
    wait_async0();                      // my async writes to LDS landed
    __syncthreads();                    // everyone's staging done, prior reads done
    const int buf = j & 1;
    if (j + 1 < jmax) stage(j + 1, buf ^ 1);   // overlap next staging with compute

    if (j*32 > qb*64 + wave*16 + 15) continue; // block fully masked for this wave

    // S = Q K^T  (16 q-rows x 32 keys), contraction over h in two 32-chunks
    f32x8 s0v = {}, s1v = {};
#pragma unroll
    for (int c = 0; c < 2; ++c) {
      const bf16x16 aqc = c ? aq1 : aq0;
      s0v = wmma_bf16(aqc, b_frag_contig(&kt[buf][n     ][c*32 + 16*half]), s0v);
      s1v = wmma_bf16(aqc, b_frag_contig(&kt[buf][n + 16][c*32 + 16*half]), s1v);
    }

    // online softmax; C-frag row = r + 8*half, col = n; row-reduce over 16 lanes
#pragma unroll
    for (int r = 0; r < 8; ++r) {
      const int qrow = qb*64 + wave*16 + r + 8*half;
      const int c0 = j*32 + n, c1 = c0 + 16;
      float x0 = (c0 <= qrow) ? s0v[r]*0.125f : -3.0e38f;   // scale = H^-0.5
      float x1 = (c1 <= qrow) ? s1v[r]*0.125f : -3.0e38f;
      float mx = fmaxf(x0, x1);
#pragma unroll
      for (int off = 1; off < 16; off <<= 1)
        mx = fmaxf(mx, __shfl_xor(mx, off, 16));
      const float mnew  = fmaxf(mrow[r], mx);
      const float alpha = __expf(mrow[r] - mnew);
      mrow[r] = mnew;
      lrow[r] *= alpha;
      o[0][r] *= alpha; o[1][r] *= alpha; o[2][r] *= alpha; o[3][r] *= alpha;
      const float p0 = __expf(x0 - mnew);
      const float p1 = __expf(x1 - mnew);
      float rs = p0 + p1;
#pragma unroll
      for (int off = 1; off < 16; off <<= 1)
        rs += __shfl_xor(rs, off, 16);
      lrow[r] += rs;
      pt[wave][r + 8*half][n]      = (__bf16)p0;   // relayout via LDS
      pt[wave][r + 8*half][16 + n] = (__bf16)p1;
    }

    // O += P V (same-wave DS store->load is in-order; V read contiguous from vt)
    const bf16x16 ap = a_frag(&pt[wave][n][0], half);
#pragma unroll
    for (int ht = 0; ht < 4; ++ht) {
      const bf16x16 bvf = b_frag_contig(&vt[buf][ht*16 + n][16*half]);
      o[ht] = wmma_bf16(ap, bvf, o[ht]);
    }
  }

  // epilogue: normalize and store f32 output
#pragma unroll
  for (int r = 0; r < 8; ++r) {
    const float inv = lrow[r] > 0.f ? 1.0f / lrow[r] : 0.f;
    const size_t row = qrow0 + r + 8*half;
#pragma unroll
    for (int ht = 0; ht < 4; ++ht)
      out[row*H_ + ht*16 + n] = o[ht][r] * inv;
  }
}

extern "C" void kernel_launch(void* const* d_in, const int* in_sizes, int n_in,
                              void* d_out, int out_size, void* d_ws, size_t ws_size,
                              hipStream_t stream) {
  (void)in_sizes; (void)n_in; (void)out_size; (void)ws_size;
  const float* x  = (const float*)d_in[0];
  const float* Wq = (const float*)d_in[1];
  const float* Wk = (const float*)d_in[2];
  const float* Wv = (const float*)d_in[3];
  float* out = (float*)d_out;

  // ws layout (bf16): WqT|WkT|WvT (3*32K) then q|k|vT (3*B*T*H) ~19.1 MB
  __bf16* wqt = (__bf16*)d_ws;
  __bf16* wkt = wqt + C_*H_;
  __bf16* wvt = wkt + C_*H_;
  __bf16* qb  = wvt + C_*H_;
  __bf16* kb  = qb + (size_t)B_*T_*H_;
  __bf16* vTb = kb + (size_t)B_*T_*H_;

  cvt_w_kernel<<<(C_*H_ + 255)/256, 256, 0, stream>>>(Wq, Wk, Wv, wqt);
  qkv_proj_kernel<<<(B_*T_)/16, 128, 0, stream>>>(x, wqt, wkt, wvt, qb, kb, vTb);
  dim3 g(T_/64, B_);
  attn_kernel<<<g, 128, 0, stream>>>(qb, kb, vTb, out);
}
